// attention_7404523618699
// MI455X (gfx1250) — matrix operand order repaired and checked by reading, not compiled, not run on hardware
//
#include <hip/hip_runtime.h>

// MHA forward for MI455X (gfx1250, wave32).
// Pre-pass: f32->f16 activation convert + weight transpose-convert.
// Then: 3x projection GEMM (all-f16 WMMA, LDS-staged weights, async-to-LDS
// when available) -> flash attention (WMMA scores + online softmax + WMMA PV)
// -> output GEMM -> f32.

#define HEADS 16
#define DKEY  64
#define DIM   1024
#define BATCH 4
#define SEQ   2048
#define NKT   (DIM / 32)   // k-steps per GEMM

typedef __attribute__((ext_vector_type(16))) _Float16 v16h;
typedef __attribute__((ext_vector_type(8)))  _Float16 v8h;
typedef __attribute__((ext_vector_type(4)))  _Float16 v4h;
typedef __attribute__((ext_vector_type(8)))  float    v8f;
typedef __attribute__((ext_vector_type(4)))  float    v4f;
typedef __attribute__((ext_vector_type(4)))  int      v4i;

#define AS1 __attribute__((address_space(1)))
#define AS3 __attribute__((address_space(3)))

// ---- async global->LDS (CDNA5) with safe fallback -------------------------
#if defined(__has_builtin)
#if __has_builtin(__builtin_amdgcn_global_load_async_to_lds_b128)
#define HAVE_ASYNC 1
#endif
#endif
#ifndef HAVE_ASYNC
#define HAVE_ASYNC 0
#endif

#if HAVE_ASYNC
#if __has_builtin(__builtin_amdgcn_s_wait_asynccnt)
#define WAIT_ASYNC(n) __builtin_amdgcn_s_wait_asynccnt(n)
#else
#define WAIT_ASYNC(n) asm volatile("s_wait_asynccnt %0" ::"i"(n) : "memory")
#endif
#endif

static __device__ inline void copy16_to_lds(_Float16* ldst,
                                            const _Float16* gsrc) {
#if HAVE_ASYNC
  // params are int4 pointers in addrspace(1)/(3) per the clang declaration
  __builtin_amdgcn_global_load_async_to_lds_b128(
      (AS1 v4i*)gsrc, (AS3 v4i*)ldst, 0, 0);
#else
  *(v8h*)ldst = *(const v8h*)gsrc;
#endif
}

static __device__ inline v8f wmma_f16(v16h a, v16h b, v8f c) {
  return __builtin_amdgcn_wmma_f32_16x16x32_f16(
      false, a, false, b, (short)0, c, false, false);
}

static __device__ inline v8f zero8() {
  v8f z;
#pragma unroll
  for (int i = 0; i < 8; ++i) z[i] = 0.0f;
  return z;
}

// ---- fragment loaders (contiguous K = 16*(l/16)+s, same for A and B) ----------------------
// A 16x32 frag: lane l%16 = row, half-group hg = l/16 -> K {hg*16+0..15}
static __device__ inline v16h load_a_f16(const _Float16* p, int ld) {
  const int lane = threadIdx.x & 31;
  const _Float16* q = p + (lane & 15) * ld + ((lane >> 4) << 4);
  v8h lo = *(const v8h*)q;
  v8h hi = *(const v8h*)(q + 8);
  v16h a;
#pragma unroll
  for (int i = 0; i < 8; ++i) { a[i] = lo[i]; a[i + 8] = hi[i]; }
  return a;
}

// B 32x16 frag where the N dim is the leading dim of an f16 row-major array:
// lane n holds row n, K contiguous (one 32B vector load).
static __device__ inline v16h load_b_f16(const _Float16* p, int ld) {
  const int lane = threadIdx.x & 31;
  return *(const v16h*)(p + (size_t)(lane & 15) * ld + ((lane >> 4) << 4));
}

// ---- pre-pass kernels -----------------------------------------------------
// elementwise f32 -> f16 (exact multiple of 1024 floats per block)
__global__ __launch_bounds__(256) void cvt_kernel(const float* __restrict__ s,
                                                  _Float16* __restrict__ d) {
  const size_t i = (size_t)blockIdx.x * 256 + threadIdx.x;
  v4f x = ((const v4f*)s)[i];
  v4h y;
#pragma unroll
  for (int j = 0; j < 4; ++j) y[j] = (_Float16)x[j];
  ((v4h*)d)[i] = y;
}

// W[k][n] f32 -> Wt[n][k] f16 via 32x32 LDS tile
__global__ __launch_bounds__(256) void wtrans_kernel(
    const float* __restrict__ W, _Float16* __restrict__ Wt) {
  __shared__ _Float16 t[32][33];
  const int k0 = (blockIdx.x & 31) * 32, n0 = (blockIdx.x >> 5) * 32;
  const int tr = threadIdx.x >> 5, tc = threadIdx.x & 31;
#pragma unroll
  for (int i = 0; i < 32; i += 8)
    t[tr + i][tc] = (_Float16)W[(size_t)(k0 + tr + i) * DIM + n0 + tc];
  __syncthreads();
#pragma unroll
  for (int i = 0; i < 32; i += 8)
    Wt[(size_t)(n0 + tr + i) * DIM + k0 + tc] = t[tc][tr + i];
}

// ---- projection GEMM: X[8192,1024]f16 @ Wt^T -> f16 -----------------------
// transpose_v == 0 : out[((b*H+h)*S + s)*64 + d]   (Q, K)
// transpose_v == 1 : out[((b*H+h)*64 + d)*S + s]   (V, transposed for PV)
__global__ __launch_bounds__(256) void proj_kernel(
    const _Float16* __restrict__ X, const _Float16* __restrict__ Wt,
    _Float16* __restrict__ out, int transpose_v) {
  __shared__ _Float16 wtile[2][64 * 32];  // double-buffered 64n x 32k tile
  const int lane  = threadIdx.x & 31;
  const int wave  = threadIdx.x >> 5;
  const int m0    = ((blockIdx.x & 63) * 8 + wave) * 16;
  const int nbase = (blockIdx.x >> 6) * 64;
  const int crow  = threadIdx.x >> 2;        // copy: 64 rows x 4 segs of 16B
  const int cseg  = (threadIdx.x & 3) * 8;

  v8f acc[4];
#pragma unroll
  for (int f = 0; f < 4; ++f) acc[f] = zero8();

  const _Float16* wsrc = Wt + (size_t)(nbase + crow) * DIM + cseg;
  copy16_to_lds(&wtile[0][crow * 32 + cseg], wsrc);

  for (int kt = 0; kt < NKT; ++kt) {
    const int cur = kt & 1;
    if (kt + 1 < NKT)
      copy16_to_lds(&wtile[cur ^ 1][crow * 32 + cseg],
                    wsrc + (kt + 1) * 32);
#if HAVE_ASYNC
    if (kt + 1 < NKT) { WAIT_ASYNC(1); } else { WAIT_ASYNC(0); }
#endif
    __syncthreads();
    v16h a = load_a_f16(X + (size_t)m0 * DIM + kt * 32, DIM);
#pragma unroll
    for (int f = 0; f < 4; ++f)
      acc[f] = wmma_f16(a, load_b_f16(&wtile[cur][f * 16 * 32], 32), acc[f]);
    __syncthreads();
  }

  const int bidx  = m0 / SEQ;
  const int s0    = m0 % SEQ;
  const int rbase = (lane >> 4) << 3;
#pragma unroll
  for (int f = 0; f < 4; ++f) {
    const int n = nbase + f * 16 + (lane & 15);
    const int h = n >> 6, d = n & 63;
#pragma unroll
    for (int r = 0; r < 8; ++r) {
      const int s = s0 + rbase + r;
      size_t off = transpose_v
          ? (((size_t)(bidx * HEADS + h) * DKEY + d) * SEQ + s)
          : (((size_t)(bidx * HEADS + h) * SEQ + s) * DKEY + d);
      out[off] = (_Float16)acc[f][r];
    }
  }
}

// ---- flash attention: one wave per 16-query tile per (b,h) ----------------
__global__ __launch_bounds__(256) void attn_kernel(
    const _Float16* __restrict__ Qh, const _Float16* __restrict__ Kh,
    const _Float16* __restrict__ Vt, _Float16* __restrict__ Ao) {
  __shared__ _Float16 pbuf[8][16 * 32];  // per-wave P tile (C->A re-stripe)
  const int lane = threadIdx.x & 31;
  const int wave = threadIdx.x >> 5;
  const int w    = blockIdx.x * 8 + wave;
  const int bh   = w >> 7;
  const int qt   = w & 127;

  const _Float16* Qbase = Qh + ((size_t)bh * SEQ + qt * 16) * DKEY;
  const v16h aq0 = load_a_f16(Qbase, DKEY);
  const v16h aq1 = load_a_f16(Qbase + 32, DKEY);

  v8f accO[4];
#pragma unroll
  for (int f = 0; f < 4; ++f) accO[f] = zero8();
  float mrow[8], lrow[8];
#pragma unroll
  for (int r = 0; r < 8; ++r) { mrow[r] = -1e30f; lrow[r] = 0.0f; }

  const float scale = 0.125f;  // 1/sqrt(64)
  _Float16* pb = pbuf[wave];
  const _Float16* Kbh = Kh + (size_t)bh * SEQ * DKEY;
  const _Float16* Vbh = Vt + (size_t)bh * DKEY * SEQ;

  for (int j = 0; j < SEQ; j += 32) {
    const _Float16* Kb0 = Kbh + (size_t)j * DKEY;
    __builtin_prefetch(Kb0 + 32 * DKEY, 0, 3);
    __builtin_prefetch(Vbh + j + 32, 0, 3);

    v8f s0 = zero8(), s1 = zero8();
    s0 = wmma_f16(aq0, load_b_f16(Kb0, DKEY), s0);
    s0 = wmma_f16(aq1, load_b_f16(Kb0 + 32, DKEY), s0);
    s1 = wmma_f16(aq0, load_b_f16(Kb0 + 16 * DKEY, DKEY), s1);
    s1 = wmma_f16(aq1, load_b_f16(Kb0 + 16 * DKEY + 32, DKEY), s1);

    float alpha[8];
#pragma unroll
    for (int r = 0; r < 8; ++r) {
      float x0 = s0[r] * scale, x1 = s1[r] * scale;
      float t = fmaxf(x0, x1);
      t = fmaxf(t, __shfl_xor(t, 1, 32));
      t = fmaxf(t, __shfl_xor(t, 2, 32));
      t = fmaxf(t, __shfl_xor(t, 4, 32));
      t = fmaxf(t, __shfl_xor(t, 8, 32));
      const float mnew = fmaxf(mrow[r], t);
      alpha[r] = __expf(mrow[r] - mnew);
      const float p0 = __expf(x0 - mnew);
      const float p1 = __expf(x1 - mnew);
      float rs = p0 + p1;
      rs += __shfl_xor(rs, 1, 32);
      rs += __shfl_xor(rs, 2, 32);
      rs += __shfl_xor(rs, 4, 32);
      rs += __shfl_xor(rs, 8, 32);
      lrow[r] = lrow[r] * alpha[r] + rs;
      mrow[r] = mnew;
      const int row = r + ((lane >> 4) << 3);
      pb[row * 32 + (lane & 15)]      = (_Float16)p0;
      pb[row * 32 + 16 + (lane & 15)] = (_Float16)p1;
    }
#pragma unroll
    for (int f = 0; f < 4; ++f)
#pragma unroll
      for (int r = 0; r < 8; ++r) accO[f][r] *= alpha[r];

    __builtin_amdgcn_wave_barrier();
    const v16h pA = load_a_f16(pb, 32);
    __builtin_amdgcn_wave_barrier();

#pragma unroll
    for (int f = 0; f < 4; ++f)
      accO[f] = wmma_f16(pA, load_b_f16(Vbh + (size_t)(f * 16) * SEQ + j, SEQ),
                         accO[f]);
  }

  const int b_ = bh >> 4, h_ = bh & 15;
#pragma unroll
  for (int f = 0; f < 4; ++f) {
    const int d = f * 16 + (lane & 15);
#pragma unroll
    for (int r = 0; r < 8; ++r) {
      const int s = qt * 16 + r + ((lane >> 4) << 3);
      Ao[(size_t)(b_ * SEQ + s) * DIM + h_ * DKEY + d] =
          (_Float16)(accO[f][r] / lrow[r]);
    }
  }
}

// ---- output projection: attn[8192,1024]f16 @ Wto^T -> f32 -----------------
__global__ __launch_bounds__(256) void outproj_kernel(
    const _Float16* __restrict__ A, const _Float16* __restrict__ Wt,
    float* __restrict__ out) {
  __shared__ _Float16 wtile[2][64 * 32];
  const int lane  = threadIdx.x & 31;
  const int wave  = threadIdx.x >> 5;
  const int m0    = ((blockIdx.x & 63) * 8 + wave) * 16;
  const int nbase = (blockIdx.x >> 6) * 64;
  const int crow  = threadIdx.x >> 2;
  const int cseg  = (threadIdx.x & 3) * 8;

  v8f acc[4];
#pragma unroll
  for (int f = 0; f < 4; ++f) acc[f] = zero8();

  const _Float16* wsrc = Wt + (size_t)(nbase + crow) * DIM + cseg;
  copy16_to_lds(&wtile[0][crow * 32 + cseg], wsrc);

  for (int kt = 0; kt < NKT; ++kt) {
    const int cur = kt & 1;
    if (kt + 1 < NKT)
      copy16_to_lds(&wtile[cur ^ 1][crow * 32 + cseg],
                    wsrc + (kt + 1) * 32);
#if HAVE_ASYNC
    if (kt + 1 < NKT) { WAIT_ASYNC(1); } else { WAIT_ASYNC(0); }
#endif
    __syncthreads();
    v16h a = load_a_f16(A + (size_t)m0 * DIM + kt * 32, DIM);
#pragma unroll
    for (int f = 0; f < 4; ++f)
      acc[f] = wmma_f16(a, load_b_f16(&wtile[cur][f * 16 * 32], 32), acc[f]);
    __syncthreads();
  }

  const int rbase = (lane >> 4) << 3;
#pragma unroll
  for (int f = 0; f < 4; ++f) {
    const int n = nbase + f * 16 + (lane & 15);
#pragma unroll
    for (int r = 0; r < 8; ++r)
      out[(size_t)(m0 + rbase + r) * DIM + n] = acc[f][r];
  }
}

extern "C" void kernel_launch(void* const* d_in, const int* in_sizes, int n_in,
                              void* d_out, int out_size, void* d_ws,
                              size_t ws_size, hipStream_t stream) {
  (void)in_sizes; (void)n_in; (void)out_size; (void)ws_size;
  const float* q_in = (const float*)d_in[0];
  const float* k_in = (const float*)d_in[1];
  const float* v_in = (const float*)d_in[2];
  const float* Wq   = (const float*)d_in[3];
  const float* Wk   = (const float*)d_in[4];
  const float* Wv   = (const float*)d_in[5];
  const float* Wo   = (const float*)d_in[6];

  _Float16* ws = (_Float16*)d_ws;
  const size_t WSZ = (size_t)DIM * DIM;                   // 1M halves
  const size_t XSZ = (size_t)BATCH * SEQ * DIM;           // 8M halves
  _Float16* Wtq = ws;
  _Float16* Wtk = ws + WSZ;
  _Float16* Wtv = ws + 2 * WSZ;
  _Float16* Wto = ws + 3 * WSZ;
  _Float16* Xb  = ws + 4 * WSZ;        // reused for q/k/v, then as Ao
  _Float16* Qh  = Xb + XSZ;
  _Float16* Kh  = Qh + XSZ;
  _Float16* Vt  = Kh + XSZ;
  _Float16* Ao  = Xb;                  // alias: X buffer is dead by then

  dim3 block(256);
  const int cvtGrid = (int)(XSZ / 4 / 256);  // 8192
  wtrans_kernel<<<dim3(1024), block, 0, stream>>>(Wq, Wtq);
  wtrans_kernel<<<dim3(1024), block, 0, stream>>>(Wk, Wtk);
  wtrans_kernel<<<dim3(1024), block, 0, stream>>>(Wv, Wtv);
  wtrans_kernel<<<dim3(1024), block, 0, stream>>>(Wo, Wto);

  cvt_kernel<<<dim3(cvtGrid), block, 0, stream>>>(q_in, Xb);
  proj_kernel<<<dim3(1024), block, 0, stream>>>(Xb, Wtq, Qh, 0);
  cvt_kernel<<<dim3(cvtGrid), block, 0, stream>>>(k_in, Xb);
  proj_kernel<<<dim3(1024), block, 0, stream>>>(Xb, Wtk, Kh, 0);
  cvt_kernel<<<dim3(cvtGrid), block, 0, stream>>>(v_in, Xb);
  proj_kernel<<<dim3(1024), block, 0, stream>>>(Xb, Wtv, Vt, 1);

  attn_kernel<<<dim3(1024), block, 0, stream>>>(Qh, Kh, Vt, Ao);
  outproj_kernel<<<dim3(1024), block, 0, stream>>>(Ao, Wto, (float*)d_out);
}
